// GlitterForConsistency_24610162606206
// MI455X (gfx1250) — matrix operand order repaired and checked by reading, not compiled, not run on hardware
//
#include <hip/hip_runtime.h>
#include <math.h>

// ---- problem structure constants (from the reference) ----
#define V          512
#define NUM_AUG    8
#define SEGS_PB    16            // segments per block  -> 128 candidates per block
#define P_STRIDE_U 260           // dwords per LDS p-row (256 data + 4 pad -> conflict-free b128)

typedef __attribute__((ext_vector_type(16))) __bf16 v16bf;
typedef __attribute__((ext_vector_type(2)))  __bf16 v2bf;
typedef __attribute__((ext_vector_type(16))) float  v16f;
typedef __attribute__((ext_vector_type(2)))  float  v2f;
typedef __attribute__((ext_vector_type(8)))  float  v8f;

__device__ __forceinline__ unsigned int pack_bf16(float a, float b) {
  // RNE f32 -> bf16 via hardware convert (v_cvt_pk_bf16_f32); even element in low half
  v2f f = { a, b };
  union { v2bf v; unsigned int u; } c;
  c.v = __builtin_convertvector(f, v2bf);
  return c.u;
}

// Fused kernel: per-segment softmax prep + WMMA dot tiles + streaming lse -> distances.
__global__ __launch_bounds__(256)
void glitter_dist_kernel(const float* __restrict__ orig,
                         const float* __restrict__ cand,
                         float* __restrict__ out_logits,   // d_out + 2B  (copy of orig)
                         float* __restrict__ distances)    // d_ws, N floats
{
  __shared__ unsigned int p_u[SEGS_PB * P_STRIDE_U];  // exp(o-m) as packed bf16 pairs
  __shared__ float dtile[8 * 256];                    // 8 waves x 16x16 f32 D tiles
  __shared__ float segscale[SEGS_PB];                 // 1/(Z*T) per local segment

  const int tid  = threadIdx.x;
  const int lane = tid & 31;
  const int w    = tid >> 5;           // wave id, 0..7
  const int bid  = blockIdx.x;

  // ---------------- phase 1: each wave preps 2 orig rows ----------------
  for (int r2 = 0; r2 < 2; ++r2) {
    const int s    = 2 * w + r2;                 // local segment 0..15
    const int gseg = bid * SEGS_PB + s;
    const float2* row = (const float2*)(orig + (size_t)gseg * V);

    float x0[8], x1[8];
    float m = -INFINITY;
#pragma unroll
    for (int j = 0; j < 8; ++j) {                // lane owns pairs (2*lane + 64j)
      float2 v = row[lane + 32 * j];
      x0[j] = v.x; x1[j] = v.y;
      m = fmaxf(m, fmaxf(v.x, v.y));
    }
#pragma unroll
    for (int off = 16; off >= 1; off >>= 1)
      m = fmaxf(m, __shfl_xor(m, off, 32));

    float zsum = 0.f;
    float2* outr = (float2*)(out_logits + (size_t)gseg * V);
#pragma unroll
    for (int j = 0; j < 8; ++j) {
      float e0 = __expf(x0[j] - m);
      float e1 = __expf(x1[j] - m);
      zsum += e0 + e1;
      p_u[s * P_STRIDE_U + lane + 32 * j] = pack_bf16(e0, e1);
      outr[lane + 32 * j] = make_float2(x0[j], x1[j]);   // selected_logits == orig row
    }
#pragma unroll
    for (int off = 16; off >= 1; off >>= 1)
      zsum += __shfl_xor(zsum, off, 32);
    if (lane == 0) segscale[s] = 1.0f / (zsum * 2.0f);   // fold 1/T into segment scale
  }
  __syncthreads();

  // ---------------- phase 2: WMMA tiles + streaming lse ----------------
  const int n = lane & 15;             // candidate column in this wave's tile
  const int h = lane >> 4;             // K-half
  const int cand_idx = bid * (SEGS_PB * NUM_AUG) + w * 16 + n;
  const float4* crow = (const float4*)(cand + (size_t)cand_idx * V);
  const unsigned int* prow = &p_u[(lane & 15) * P_STRIDE_U];

  v8f acc = {};
  float M = -INFINITY, S = 0.f;        // online logsumexp of cand/T (per lane half-column)

#pragma unroll 4                        // moderate unroll: keep VGPRs < 256, 16 loads in flight
  for (int t = 0; t < 16; ++t) {
    // ---- B fragment: 16 contiguous f32 of this lane's candidate column ----
    const int kb = (32 * t + 16 * h) >> 2;   // float4 index
    float4 q0 = crow[kb + 0], q1 = crow[kb + 1], q2 = crow[kb + 2], q3 = crow[kb + 3];
    v16f xf;
    xf[0]  = q0.x; xf[1]  = q0.y; xf[2]  = q0.z; xf[3]  = q0.w;
    xf[4]  = q1.x; xf[5]  = q1.y; xf[6]  = q1.z; xf[7]  = q1.w;
    xf[8]  = q2.x; xf[9]  = q2.y; xf[10] = q2.z; xf[11] = q2.w;
    xf[12] = q3.x; xf[13] = q3.y; xf[14] = q3.z; xf[15] = q3.w;

    // one packed hardware convert per dword pair (v_cvt_pk_bf16_f32 x8)
    v16bf bfv = __builtin_convertvector(xf, v16bf);

    // ---- streaming lse on the same values (scaled by 1/T = 0.5) ----
    float mx = fmaxf(xf[0], xf[1]);
#pragma unroll
    for (int j = 2; j < 16; ++j) mx = fmaxf(mx, xf[j]);
    mx *= 0.5f;
    float cs = 0.f;
#pragma unroll
    for (int j = 0; j < 16; ++j) cs += __expf(xf[j] * 0.5f - mx);
    float nM = fmaxf(M, mx);
    S = S * __expf(M - nM) + cs * __expf(mx - nM);
    M = nM;

    // ---- A fragment from LDS: p row (lane&15), K = 32t + 16h + {0..15} ----
    const int ka = 16 * t + 8 * h;           // dword index into p row
    union { v16bf v; uint4 q[2]; } af;
    af.q[0] = *(const uint4*)&prow[ka];
    af.q[1] = *(const uint4*)&prow[ka + 4];

    acc = __builtin_amdgcn_wmma_f32_16x16x32_bf16(
        /*neg_a=*/false, af.v, /*neg_b=*/false, bfv,
        /*c_mod=*/(short)0, acc, /*reuse_a=*/false, /*reuse_b=*/false);
  }

  // stage D tile so any lane can grab its block-diagonal entry
  union { v8f v; float f[8]; } au; au.v = acc;
#pragma unroll
  for (int r = 0; r < 8; ++r)
    dtile[w * 256 + r * 32 + lane] = au.f[r];

  // merge the two half-column lse partials (lanes l and l^16)
  float oM = __shfl_xor(M, 16, 32);
  float oS = __shfl_xor(S, 16, 32);
  float fM = fmaxf(M, oM);
  float fS = S * __expf(M - fM) + oS * __expf(oM - fM);

  __syncthreads();

  if (lane < 16) {
    const int mrow = 2 * w + (lane >> 3);            // candidate's own local segment
    const float u  = dtile[w * 256 + (mrow & 7) * 32 + lane + 16 * (mrow >> 3)];
    const float lse = __logf(fS) + fM;
    // dist = lse(c/T) - dot(p, c)/T   (per-segment constant H dropped: rank-equivalent)
    distances[cand_idx] = lse - u * segscale[mrow];
  }
}

// Iterative per-segment argmax (augment_rank rounds, ties -> lowest index).
__global__ __launch_bounds__(256)
void glitter_select_kernel(const float* __restrict__ distances,
                           float* __restrict__ out, int B,
                           const int* __restrict__ rank_ptr)
{
  const int s = blockIdx.x * blockDim.x + threadIdx.x;
  if (s >= B) return;
  const float4* p = (const float4*)(distances + (size_t)s * NUM_AUG);
  float4 a = p[0], b = p[1];
  float d[NUM_AUG] = { a.x, a.y, a.z, a.w, b.x, b.y, b.z, b.w };

  int rank = rank_ptr[0];
  if (rank < 1) rank = 1;
  int bj = 0;
  for (int r = 0; r < rank; ++r) {
    float best = -INFINITY; bj = 0;
#pragma unroll
    for (int j = 0; j < NUM_AUG; ++j)
      if (d[j] > best) { best = d[j]; bj = j; }
    d[bj] = -INFINITY;
  }
  out[s]     = (float)(s * NUM_AUG + bj);   // selected candidate index
  out[B + s] = (float)bj;                   // selected rank (idx % NUM_AUG)
}

extern "C" void kernel_launch(void* const* d_in, const int* in_sizes, int n_in,
                              void* d_out, int out_size, void* d_ws, size_t ws_size,
                              hipStream_t stream) {
  const float* orig = (const float*)d_in[0];
  const float* cand = (const float*)d_in[1];
  const int*   rank = (const int*)d_in[4];     // augment_rank (device scalar)

  const int B = in_sizes[0] / V;               // 8192
  float* out        = (float*)d_out;
  float* out_logits = out + 2 * (size_t)B;     // [selected | ranks | logits]
  float* distances  = (float*)d_ws;            // N floats

  glitter_dist_kernel<<<B / SEGS_PB, 256, 0, stream>>>(orig, cand, out_logits, distances);
  glitter_select_kernel<<<(B + 255) / 256, 256, 0, stream>>>(distances, out, B, rank);
}
